// Dots1Attention_48636209660327
// MI455X (gfx1250) — matrix operand order repaired and checked by reading, not compiled, not run on hardware
//
#include <hip/hip_runtime.h>
#include <hip/hip_bf16.h>
#include <cstdint>

// ---------------------------------------------------------------------------
// Types for CDNA5 WMMA (gfx1250, wave32): 16x16x32 f16 -> f32
// ---------------------------------------------------------------------------
typedef __attribute__((ext_vector_type(16))) _Float16 v16h;
typedef __attribute__((ext_vector_type(8)))  float    v8f;
typedef __attribute__((ext_vector_type(4)))  float    vf4;

union F16Frag { v16h v; vf4 q[2]; };

__device__ __forceinline__ v8f wmma32(v16h a, v16h b, v8f c) {
    // D = A(16x32,f16) * B(32x16,f16) + C(16x16,f32)
    return __builtin_amdgcn_wmma_f32_16x16x32_f16(
        /*neg_a=*/false, a, /*neg_b=*/false, b,
        /*c_mod=*/(short)0, c, /*reuse_a=*/false, /*reuse_b=*/false);
}

// Async global->LDS copy of 16 bytes per lane (GLOBAL_LOAD_ASYNC_TO_LDS_B128,
// tracked by ASYNCcnt). lds_off is the 32-bit LDS byte address (generic ptr
// truncated to its low 32 bits).
__device__ __forceinline__ void async_copy_b128(unsigned lds_off, unsigned long long gaddr) {
    asm volatile("global_load_async_to_lds_b128 %0, %1, off"
                 :: "v"(lds_off), "v"(gaddr) : "memory");
}
__device__ __forceinline__ void wait_async0() {
    asm volatile("s_wait_asynccnt 0" ::: "memory");
}
__device__ __forceinline__ unsigned lds_addr32(const void* p) {
    return (unsigned)(unsigned long long)p;
}

// A-matrix 16x32 (MxK), 16-bit (ISA 7.12.2): lane L holds row M=L%16.
// Half h maps to K = (h/8)*16 + (L/16)*8 + (h%8)  -> two contiguous 16B chunks.
__device__ __forceinline__ v16h load_frag_a(const _Float16* base, int ld, int lane) {
    const int m  = lane & 15;
    const int hi = lane >> 4;
    const _Float16* row = base + (size_t)m * ld;
    F16Frag f;
    f.q[0] = *(const vf4*)(row + hi * 8);
    f.q[1] = *(const vf4*)(row + 16 + hi * 8);
    return f.v;
}

// B-matrix 32x16 (KxN), 16-bit, staged as [N][K] (K-major rows): lane L holds
// column N=L%16; half h maps to K = (h/8)*16 + (L/16)*8 + (h%8) -> two 16B chunks.
__device__ __forceinline__ v16h load_frag_b(const _Float16* base, int ld, int lane) {
    const int n  = lane & 15;
    const int hi = lane >> 4;
    const _Float16* row = base + (size_t)n * ld;
    F16Frag f;
    f.q[0] = *(const vf4*)(row + hi * 8);
    f.q[1] = *(const vf4*)(row + 16 + hi * 8);
    return f.v;
}

// Row statistics live in a 16-lane half (C-layout: row = vgpr + 8*(lane/16)).
__device__ __forceinline__ float rowmax16(float x) {
    x = fmaxf(x, __shfl_xor(x, 1, 32));
    x = fmaxf(x, __shfl_xor(x, 2, 32));
    x = fmaxf(x, __shfl_xor(x, 4, 32));
    x = fmaxf(x, __shfl_xor(x, 8, 32));
    return x;
}
__device__ __forceinline__ float rowsum16(float x) {
    x += __shfl_xor(x, 1, 32);
    x += __shfl_xor(x, 2, 32);
    x += __shfl_xor(x, 4, 32);
    x += __shfl_xor(x, 8, 32);
    return x;
}

// ---------------------------------------------------------------------------
// Bulk f32 -> f16 conversion (8 elems/thread, fully coalesced)
// ---------------------------------------------------------------------------
__global__ __launch_bounds__(256) void cvt_f32_f16(const float* __restrict__ src,
                                                   _Float16* __restrict__ dst, int n) {
    const int i = (blockIdx.x * 256 + threadIdx.x) * 8;
    if (i >= n) return;
    vf4 a = *(const vf4*)(src + i);
    vf4 b = *(const vf4*)(src + i + 4);
    _Float16* d = dst + i;
    d[0] = (_Float16)a.x; d[1] = (_Float16)a.y; d[2] = (_Float16)a.z; d[3] = (_Float16)a.w;
    d[4] = (_Float16)b.x; d[5] = (_Float16)b.y; d[6] = (_Float16)b.z; d[7] = (_Float16)b.w;
}

// ---------------------------------------------------------------------------
// C[M,N] = A[M,K] * B[N,K]^T   (f16 in, f32 WMMA accum, f32 out)
// 128x128 block tile, 8 waves = 2(M) x 4(N), K step 32.
// Double-buffered LDS, staged with async global->LDS B128 copies.
// ---------------------------------------------------------------------------
__global__ __launch_bounds__(256) void wmma_gemm_nt_f16(const _Float16* __restrict__ A,
                                                        const _Float16* __restrict__ B,
                                                        float* __restrict__ C,
                                                        int M, int N, int K) {
    __shared__ _Float16 As[2][128 * 32];
    __shared__ _Float16 Bs[2][128 * 32];

    const int tid  = threadIdx.x;
    const int lane = tid & 31;
    const int wave = tid >> 5;
    const int wm   = wave >> 2;   // 0..1 : 64 rows
    const int wn   = wave & 3;    // 0..3 : 32 cols
    const int bm   = blockIdx.x * 128;
    const int bn   = blockIdx.y * 128;

    v8f acc[4][2] = {};

    // Each thread owns 16 halves (32B) of each 8KB tile: linear off = tid*16
    // halves == row (tid>>1), col (tid&1)*16 in the [128][32] layout.
    const int srow = tid >> 1;
    const int scol = (tid & 1) * 16;
    const _Float16* aSrc = A + (size_t)(bm + srow) * K + scol;
    const _Float16* bSrc = B + (size_t)(bn + srow) * K + scol;
    const unsigned ldsA0 = lds_addr32(&As[0][tid * 16]);
    const unsigned ldsB0 = lds_addr32(&Bs[0][tid * 16]);

    auto stage = [&](int buf, int k0) {
        const unsigned la = ldsA0 + (unsigned)buf * (128 * 32 * 2);
        const unsigned lb = ldsB0 + (unsigned)buf * (128 * 32 * 2);
        async_copy_b128(la,      (unsigned long long)(aSrc + k0));
        async_copy_b128(la + 16, (unsigned long long)(aSrc + k0 + 8));
        async_copy_b128(lb,      (unsigned long long)(bSrc + k0));
        async_copy_b128(lb + 16, (unsigned long long)(bSrc + k0 + 8));
    };

    stage(0, 0);
    wait_async0();
    __syncthreads();

    const int nk = K / 32;
    for (int ki = 0; ki < nk; ++ki) {
        const int buf = ki & 1;
        if (ki + 1 < nk) stage(buf ^ 1, (ki + 1) * 32);   // overlap copy w/ WMMA

        v16h bf[2];
#pragma unroll
        for (int j = 0; j < 2; ++j)
            bf[j] = load_frag_b(&Bs[buf][(wn * 32 + j * 16) * 32], 32, lane);
#pragma unroll
        for (int i = 0; i < 4; ++i) {
            v16h af = load_frag_a(&As[buf][(wm * 64 + i * 16) * 32], 32, lane);
#pragma unroll
            for (int j = 0; j < 2; ++j)
                acc[i][j] = wmma32(af, bf[j], acc[i][j]);
        }
        wait_async0();
        __syncthreads();
    }

    // C layout: vgpr rr -> row rr + 8*(lane/16); col = lane%16
    const int cm  = lane & 15;
    const int chi = lane >> 4;
#pragma unroll
    for (int i = 0; i < 4; ++i)
#pragma unroll
        for (int j = 0; j < 2; ++j)
#pragma unroll
            for (int rr = 0; rr < 8; ++rr) {
                const int row = bm + wm * 64 + i * 16 + rr + 8 * chi;
                const int col = bn + wn * 32 + j * 16 + cm;
                C[(size_t)row * N + col] = acc[i][j][rr];
            }
}

// ---------------------------------------------------------------------------
// Per-(token, head) RMSNorm + RoPE, emit f16 q/k/v. One wave per task.
// tasks: c in [0,32) -> q head, [32,40) -> k head, [40,48) -> v head.
// ---------------------------------------------------------------------------
__global__ __launch_bounds__(256) void rmsnorm_rope_cvt(const float* __restrict__ qkv,
                                                        const float* __restrict__ qw,
                                                        const float* __restrict__ kw,
                                                        _Float16* __restrict__ qh,
                                                        _Float16* __restrict__ kh,
                                                        _Float16* __restrict__ vh,
                                                        int T) {
    const int lane = threadIdx.x & 31;
    const int wave = threadIdx.x >> 5;
    const int task = blockIdx.x * 8 + wave;
    const int t = task / 48;
    const int c = task % 48;
    if (t >= T) return;

    const float* src = qkv + (size_t)t * 6144 + c * 128;

    if (c >= 40) {   // V: plain f32 -> f16 copy
        _Float16* dst = vh + ((size_t)t * 8 + (c - 40)) * 128;
#pragma unroll
        for (int i = 0; i < 4; ++i)
            dst[lane + 32 * i] = (_Float16)src[lane + 32 * i];
        return;
    }

    const bool  isq = (c < 32);
    const float* w  = isq ? qw : kw;

    float x0 = src[lane];
    float x1 = src[lane + 32];
    float x2 = src[lane + 64];
    float x3 = src[lane + 96];

    // RMS norm over 128 dims
    float ss = x0 * x0 + x1 * x1 + x2 * x2 + x3 * x3;
#pragma unroll
    for (int msk = 1; msk < 32; msk <<= 1) ss += __shfl_xor(ss, msk, 32);
    const float rinv = rsqrtf(ss * (1.0f / 128.0f) + 1e-6f);
    x0 = x0 * rinv * w[lane];
    x1 = x1 * rinv * w[lane + 32];
    x2 = x2 * rinv * w[lane + 64];
    x3 = x3 * rinv * w[lane + 96];

    // RoPE: pairs (i, i+64); lane handles i=lane and i=lane+32
    const float pos  = (float)t;                  // positions == arange(T)
    const float l2th = 13.287712379549449f;       // log2(10000)
    const float inv0 = exp2f(-(float)lane        * (l2th / 64.0f));
    const float inv1 = exp2f(-(float)(lane + 32) * (l2th / 64.0f));
    float s0, c0, s1, c1;
    sincosf(pos * inv0, &s0, &c0);
    sincosf(pos * inv1, &s1, &c1);
    const float o0 = x0 * c0 - x2 * s0;
    const float o2 = x2 * c0 + x0 * s0;
    const float o1 = x1 * c1 - x3 * s1;
    const float o3 = x3 * c1 + x1 * s1;

    _Float16* dst = isq ? (qh + ((size_t)t * 32 + c) * 128)
                        : (kh + ((size_t)t * 8 + (c - 32)) * 128);
    dst[lane]      = (_Float16)o0;
    dst[lane + 32] = (_Float16)o1;
    dst[lane + 64] = (_Float16)o2;
    dst[lane + 96] = (_Float16)o3;
}

// ---------------------------------------------------------------------------
// Causal flash attention. Block = (128 queries, 1 head), 8 waves * 16 queries.
// Key tiles of 32; S = Q*K^T and O += P*V via WMMA; online softmax with
// 16-lane shuffles; P repacked to an A-fragment through per-wave LDS scratch.
// Output written directly as f16 for the WO projection.
// ---------------------------------------------------------------------------
__global__ __launch_bounds__(256) void flash_attn(const _Float16* __restrict__ qh,
                                                  const _Float16* __restrict__ kh,
                                                  const _Float16* __restrict__ vh,
                                                  _Float16* __restrict__ out) {
    __shared__ _Float16 Ks[32 * 128];     // [key][d]
    __shared__ _Float16 Vts[128 * 32];    // [d][key]  (transposed V)
    __shared__ _Float16 Ps[8][16 * 32];   // per-wave P scratch [M][K]

    const int tid  = threadIdx.x;
    const int lane = tid & 31;
    const int wave = tid >> 5;
    const int h    = blockIdx.y;
    const int q0   = blockIdx.x * 128 + wave * 16;
    const int kvh  = h >> 2;              // GQA group of 4
    const int m    = lane & 15;
    const int hi   = lane >> 4;

    // Q fragments: 16 queries x 128 dims, row stride = 32 heads * 128
    v16h qf[4];
    const _Float16* qbase = qh + ((size_t)q0 * 32 + h) * 128;
#pragma unroll
    for (int d = 0; d < 4; ++d)
        qf[d] = load_frag_a(qbase + d * 32, 32 * 128, lane);

    v8f   o[8] = {};
    float mr[8], lr[8];
#pragma unroll
    for (int rr = 0; rr < 8; ++rr) { mr[rr] = -1e30f; lr[rr] = 0.0f; }

    const int   nkb       = (blockIdx.x * 128 + 127) / 32 + 1;
    const int   key_stage = tid >> 3;        // 0..31
    const int   d_stage   = (tid & 7) * 16;  // 16 dims per thread
    const float scale     = 0.088388347648318447f;   // 1/sqrt(128)

    for (int kb = 0; kb < nkb; ++kb) {
        const int kbase = kb * 32;
        {   // cooperative K / V^T staging
            const _Float16* ksrc = kh + ((size_t)(kbase + key_stage) * 8 + kvh) * 128 + d_stage;
            *(vf4*)(Ks + key_stage * 128 + d_stage)     = *(const vf4*)ksrc;
            *(vf4*)(Ks + key_stage * 128 + d_stage + 8) = *(const vf4*)(ksrc + 8);
            const _Float16* vsrc = vh + ((size_t)(kbase + key_stage) * 8 + kvh) * 128 + d_stage;
#pragma unroll
            for (int i = 0; i < 16; ++i)
                Vts[(d_stage + i) * 32 + key_stage] = vsrc[i];
        }
        __syncthreads();

        if (kbase <= q0 + 15) {           // wave has unmasked keys in this tile
            v8f s[2] = {};
#pragma unroll
            for (int d = 0; d < 4; ++d) { // K^T: B[n=key][k=d] == Ks rows
                v16h kf0 = load_frag_b(Ks + d * 32, 128, lane);
                v16h kf1 = load_frag_b(Ks + 16 * 128 + d * 32, 128, lane);
                s[0] = wmma32(qf[d], kf0, s[0]);
                s[1] = wmma32(qf[d], kf1, s[1]);
            }

            _Float16* ps = &Ps[wave][0];
#pragma unroll
            for (int rr = 0; rr < 8; ++rr) {
                const int qrow = q0 + rr + 8 * hi;
                float v0 = s[0][rr] * scale;
                float v1 = s[1][rr] * scale;
                if (kbase + m > qrow)      v0 = -1e30f;   // causal mask
                if (kbase + 16 + m > qrow) v1 = -1e30f;
                const float mx = rowmax16(fmaxf(v0, v1));
                const float mn = fmaxf(mr[rr], mx);
                const float p0 = __expf(v0 - mn);
                const float p1 = __expf(v1 - mn);
                const float rs = rowsum16(p0 + p1);
                const float alpha = __expf(mr[rr] - mn);
                mr[rr] = mn;
                lr[rr] = lr[rr] * alpha + rs;
#pragma unroll
                for (int t = 0; t < 8; ++t) o[t][rr] *= alpha;
                ps[(rr + 8 * hi) * 32 + m]      = (_Float16)p0;
                ps[(rr + 8 * hi) * 32 + 16 + m] = (_Float16)p1;
            }

            // O += P(16x32) * V(32x128): B[n=d][k=key] == Vts rows
            v16h pf = load_frag_a(ps, 32, lane);
#pragma unroll
            for (int t = 0; t < 8; ++t) {
                v16h vf = load_frag_b(Vts + (t * 16) * 32, 32, lane);
                o[t] = wmma32(pf, vf, o[t]);
            }
        }
        __syncthreads();
    }

    // epilogue: normalize, write f16 [t][h*128+d]
#pragma unroll
    for (int rr = 0; rr < 8; ++rr) {
        const float inv = 1.0f / lr[rr];
        const int   row = q0 + rr + 8 * hi;
        _Float16* dst = out + (size_t)row * 4096 + h * 128;
#pragma unroll
        for (int t = 0; t < 8; ++t)
            dst[t * 16 + m] = (_Float16)(o[t][rr] * inv);
    }
}

// ---------------------------------------------------------------------------
// Launcher
// inputs: 0 positions (arange, unused), 1 hidden [T,4096] f32,
//         2 wqkv [6144,4096] f32, 3 wo [4096,4096] f32, 4 q_norm_w, 5 k_norm_w
// ---------------------------------------------------------------------------
extern "C" void kernel_launch(void* const* d_in, const int* in_sizes, int n_in,
                              void* d_out, int out_size, void* d_ws, size_t ws_size,
                              hipStream_t stream) {
    const float* hidden = (const float*)d_in[1];
    const float* wqkv   = (const float*)d_in[2];
    const float* wo     = (const float*)d_in[3];
    const float* qw     = (const float*)d_in[4];
    const float* kw     = (const float*)d_in[5];
    float* out = (float*)d_out;

    const int T       = in_sizes[0];        // 2048
    const int QKV_OUT = 48 * 128;           // 6144
    const int HID     = 4096;

    char* ws = (char*)d_ws;
    size_t off = 0;
    float*    qkv = (float*)(ws + off);      off += (size_t)T * QKV_OUT * 4;   // f32 qkv
    _Float16* qhx = (_Float16*)(ws + off);   off += (size_t)T * 32 * 128 * 2;
    _Float16* khx = (_Float16*)(ws + off);   off += (size_t)T * 8 * 128 * 2;
    _Float16* vhx = (_Float16*)(ws + off);   off += (size_t)T * 8 * 128 * 2;
    _Float16* hh  = (_Float16*)(ws + off);   off += (size_t)T * HID * 2;       // hidden f16
    _Float16* wqh = (_Float16*)(ws + off);   off += (size_t)QKV_OUT * HID * 2; // wqkv f16
    _Float16* woh = (_Float16*)(ws + off);   off += (size_t)HID * HID * 2;     // wo f16
    _Float16* attnh = hh;   // reuse hidden-f16 buffer after QKV GEMM consumed it

    // 0) one-shot f32 -> f16 conversions (bandwidth-trivial at 23.3 TB/s)
    cvt_f32_f16<<<(T * HID) / 2048, 256, 0, stream>>>(hidden, hh, T * HID);
    cvt_f32_f16<<<(QKV_OUT * HID) / 2048, 256, 0, stream>>>(wqkv, wqh, QKV_OUT * HID);
    cvt_f32_f16<<<(HID * HID) / 2048, 256, 0, stream>>>(wo, woh, HID * HID);

    // 1) QKV projection: [T,4096] @ [6144,4096]^T -> f32
    wmma_gemm_nt_f16<<<dim3(T / 128, QKV_OUT / 128), 256, 0, stream>>>(
        hh, wqh, qkv, T, QKV_OUT, HID);

    // 2) RMSNorm + RoPE + f16 conversion (one wave per (t, head))
    rmsnorm_rope_cvt<<<(T * 48) / 8, 256, 0, stream>>>(
        qkv, qw, kw, qhx, khx, vhx, T);

    // 3) Causal flash attention (32 heads x T/128 query blocks) -> f16
    flash_attn<<<dim3(T / 128, 32), 256, 0, stream>>>(qhx, khx, vhx, attnh);

    // 4) Output projection: [T,4096] @ [4096,4096]^T -> d_out (f32)
    wmma_gemm_nt_f16<<<dim3(T / 128, HID / 128), 256, 0, stream>>>(
        attnh, woh, out, T, HID, HID);
}
